// Model_2817498546516
// MI455X (gfx1250) — matrix operand order repaired and checked by reading, not compiled, not run on hardware
//
#include <hip/hip_runtime.h>

// ---------------------------------------------------------------------------
// Types
// ---------------------------------------------------------------------------
typedef __attribute__((ext_vector_type(16))) _Float16     v16h;
typedef __attribute__((ext_vector_type(8)))  float        v8f;
typedef __attribute__((ext_vector_type(4)))  float        f32x4;
typedef __attribute__((ext_vector_type(4)))  unsigned int u32x4;
typedef __attribute__((ext_vector_type(8)))  int          i32x8;
typedef __attribute__((ext_vector_type(4)))  int          i32x4;

// ---------------------------------------------------------------------------
// Problem constants (match reference)
// ---------------------------------------------------------------------------
#define NPTS      131072      // 4096 rays * 32 samples
#define NRAYS     4096
#define BM        128         // points per workgroup in MLP kernel
#define NWAVES    4
#define NTHREADS  128

// Grid-encode level metadata: scale = 16*2^l - 1, res = 16<<l,
// dense for l<=3 (res^3 <= 2^21), all sizes are powers of two.
__constant__ int LVL_OFF[10] = {
    0, 4096, 36864, 299008, 2396160, 4493312,
    6590464, 8687616, 10784768, 12881920
};

// Workspace layout (in halves)
#define OFF_WD0 0        // Kpad=64,  N=64   -> 4096 halves
#define OFF_WD1 4096     // Kpad=64,  N=256  -> 16384
#define OFF_W0  20480    // Kpad=288, N=256  -> 73728
#define OFF_W1  94208    // Kpad=544, N=256  -> 139264
#define OFF_WR  233472   // Kpad=256, N=16   -> 4096
#define OFF_FEATS 237568                  // NPTS x 64 f16 (16 MB)
#define OFF_DE    (OFF_FEATS + NPTS*64)   // NRAYS x 32 f16 (256 KB)

// LDS layout (halves)
#define H0_OFF   (128*64)            // sFeat: [128][64] at 0
#define INP_OFF  (H0_OFF + 128*64)   // [128][288] (x[256] + dir_enc[27] + pad)
#define H1_OFF   (INP_OFF + 128*288) // [128][256]
#define H2_OFF   (H1_OFF + 128*256)  // [128][256]
#define LDS_HALVES (H2_OFF + 128*256)
#define SMEM_BYTES (LDS_HALVES*2 + BM*4)   // + f32 density column

// ---------------------------------------------------------------------------
// Weight packing: f32 (K x N row-major) -> f16 WMMA B-fragment blocks.
// Block (nt, ks) holds a 32(K) x 16(N) tile as 32 lanes x 16 halves:
//   lane L: column n = nt*16 + (L&15); halves = K rows ks*32 + (L>>4)*8 + {0..7, 16..23}
// ---------------------------------------------------------------------------
__global__ __launch_bounds__(256)
void pack_weights(const float* __restrict__ W, int Kreal, int Nreal,
                  int Kpad, int Npad, _Float16* __restrict__ dst) {
    int tid = blockIdx.x * 256 + threadIdx.x;
    int total = Kpad * Npad;
    if (tid >= total) return;
    int inlane = tid & 15;          // K offset within the lane's chunk
    int lane   = (tid >> 4) & 31;
    int blk    = tid >> 9;          // which 32x16 block
    int ksteps = Kpad >> 5;
    int ks = blk % ksteps;
    int nt = blk / ksteps;
    int n  = nt * 16 + (lane & 15);
    int kk = ks * 32 + (lane >> 4) * 8 + (inlane & 7) + (inlane >> 3) * 16;
    float v = (kk < Kreal && n < Nreal) ? W[(size_t)kk * Nreal + n] : 0.0f;
    dst[tid] = (_Float16)v;
}

// ---------------------------------------------------------------------------
// High-occupancy grid-encode kernel: contract + 10-level hash-grid gather.
// One thread per point, no LDS -> deep wave occupancy to hide HBM latency
// of the ~80 random float4 gathers per point (table is 240MB > 192MB L2).
// Writes 64 f16 (40 feats + zero pad) per point to workspace.
// ---------------------------------------------------------------------------
__global__ __launch_bounds__(256)
void grid_encode_kernel(const float* __restrict__ coords,
                        const float* __restrict__ table,
                        _Float16* __restrict__ feats) {
    int p = blockIdx.x * 256 + threadIdx.x;   // NPTS % 256 == 0
    float cx = coords[(size_t)p * 3 + 0];
    float cy = coords[(size_t)p * 3 + 1];
    float cz = coords[(size_t)p * 3 + 2];
    float n2  = cx * cx + cy * cy + cz * cz;
    float nrm = sqrtf(fmaxf(n2, 1e-12f));
    if (nrm > 1.0f) {
        float s = (2.0f - 1.0f / nrm) / nrm;
        cx *= s; cy *= s; cz *= s;
    }
    // c = contract/2 ; x01 = (c+1)/2  => x01 = contract*0.25 + 0.5
    float x0 = cx * 0.25f + 0.5f;
    float y0 = cy * 0.25f + 0.5f;
    float z0 = cz * 0.25f + 0.5f;

    _Float16 buf[64];
    const f32x4* tab4 = (const f32x4*)table;
#pragma unroll
    for (int l = 0; l < 10; ++l) {
        const unsigned res = 16u << l;
        const float scale  = (float)res - 1.0f;
        const bool dense   = (l <= 3);
        const unsigned mask = dense ? (res * res * res - 1u) : ((1u << 21) - 1u);
        const int off = LVL_OFF[l];
        float px = x0 * scale + 0.5f;
        float py = y0 * scale + 0.5f;
        float pz = z0 * scale + 0.5f;
        float fx0 = floorf(px), fy0 = floorf(py), fz0 = floorf(pz);
        float fx = px - fx0, fy = py - fy0, fz = pz - fz0;
        unsigned ix = (unsigned)(int)fx0;
        unsigned iy = (unsigned)(int)fy0;
        unsigned iz = (unsigned)(int)fz0;
        float a0 = 0.f, a1 = 0.f, a2 = 0.f, a3 = 0.f;
#pragma unroll
        for (int c8 = 0; c8 < 8; ++c8) {
            const int dx = (c8 >> 2) & 1, dy = (c8 >> 1) & 1, dz = c8 & 1;
            unsigned gx = ix + (unsigned)dx;
            unsigned gy = iy + (unsigned)dy;
            unsigned gz = iz + (unsigned)dz;
            float w = (dx ? fx : 1.0f - fx) *
                      (dy ? fy : 1.0f - fy) *
                      (dz ? fz : 1.0f - fz);
            unsigned idx;
            if (dense) idx = (gx * res + gy) * res + gz;
            else idx = gx ^ (gy * 2654435761u) ^ (gz * 805459861u);
            idx &= mask;
            f32x4 r = tab4[(unsigned)off + idx];
            a0 += w * r.x; a1 += w * r.y; a2 += w * r.z; a3 += w * r.w;
        }
        buf[l * 4 + 0] = (_Float16)a0;
        buf[l * 4 + 1] = (_Float16)a1;
        buf[l * 4 + 2] = (_Float16)a2;
        buf[l * 4 + 3] = (_Float16)a3;
    }
#pragma unroll
    for (int j = 40; j < 64; ++j) buf[j] = (_Float16)0.f;

    f32x4* dst = (f32x4*)(feats + (size_t)p * 64);
    const f32x4* src = (const f32x4*)buf;
#pragma unroll
    for (int i = 0; i < 8; ++i) dst[i] = src[i];
}

// ---------------------------------------------------------------------------
// Per-ray direction encoding: [d(3), sin(d*2^i)(12), cos(d*2^i)(12), pad(5)]
// ---------------------------------------------------------------------------
__global__ __launch_bounds__(256)
void direnc_kernel(const float* __restrict__ viewdirs,
                   _Float16* __restrict__ de) {
    int r = blockIdx.x * 256 + threadIdx.x;
    if (r >= NRAYS) return;
    float dx = viewdirs[(size_t)r * 3 + 0];
    float dy = viewdirs[(size_t)r * 3 + 1];
    float dz = viewdirs[(size_t)r * 3 + 2];
    _Float16 buf[32];
    buf[0] = (_Float16)dx; buf[1] = (_Float16)dy; buf[2] = (_Float16)dz;
#pragma unroll
    for (int i = 0; i < 4; ++i) {
        float s = (float)(1 << i);
        buf[3 + i * 3 + 0]  = (_Float16)sinf(dx * s);
        buf[3 + i * 3 + 1]  = (_Float16)sinf(dy * s);
        buf[3 + i * 3 + 2]  = (_Float16)sinf(dz * s);
        buf[15 + i * 3 + 0] = (_Float16)cosf(dx * s);
        buf[15 + i * 3 + 1] = (_Float16)cosf(dy * s);
        buf[15 + i * 3 + 2] = (_Float16)cosf(dz * s);
    }
#pragma unroll
    for (int j = 27; j < 32; ++j) buf[j] = (_Float16)0.f;
    f32x4* dst = (f32x4*)(de + (size_t)r * 32);
    const f32x4* src = (const f32x4*)buf;
#pragma unroll
    for (int i = 0; i < 4; ++i) dst[i] = src[i];
}

// ---------------------------------------------------------------------------
// WMMA helpers
// ---------------------------------------------------------------------------
// A fragment (16-bit A, 16x32): lane L holds row m = mt*16 + (L&15);
// lanes 0-15 hold K = {0..7, 16..23}, lanes 16-31 hold K = {8..15, 24..31}
// (relative to ks*32). Two contiguous 16-byte LDS reads per fragment.
__device__ __forceinline__ v16h load_a_frag(const _Float16* A, int Kpad,
                                            int m, int hi, int ks) {
    union { v16h h; f32x4 v[2]; } u;
    const _Float16* p = A + (size_t)m * Kpad + (ks << 5) + (hi << 3);
    u.v[0] = *(const f32x4*)p;
    u.v[1] = *(const f32x4*)(p + 16);
    return u.h;
}

__device__ __forceinline__ v8f mm_accum(v8f acc, const _Float16* A, int Kpad,
                                        int m, int hi, const v16h* B,
                                        int bBase, int ksN, int lane) {
#pragma unroll
    for (int ks = 0; ks < ksN; ++ks) {
        v16h a = load_a_frag(A, Kpad, m, hi, ks);
        v16h b = B[(size_t)(bBase + ks) * 32 + lane];
        acc = __builtin_amdgcn_wmma_f32_16x16x32_f16(
            false, a, false, b, (short)0, acc, false, false);
    }
    return acc;
}

// ---------------------------------------------------------------------------
// Fused MLP kernel: one workgroup = 128 points. Feature tile is DMA'd
// global->LDS by the Tensor Data Mover; all GEMMs are f16 WMMA, f32 accum.
// ---------------------------------------------------------------------------
__global__ __launch_bounds__(NTHREADS)
void nerf_mlp(const _Float16* __restrict__ feats,
              const _Float16* __restrict__ deG,
              const _Float16* __restrict__ wpack,
              const float* __restrict__ bd0, const float* __restrict__ bd1,
              const float* __restrict__ b0,  const float* __restrict__ b1,
              const float* __restrict__ br,
              float* __restrict__ out) {
    extern __shared__ __align__(16) unsigned char smem_raw[];
    _Float16* sFeat = (_Float16*)smem_raw;          // [128][64] at LDS offset 0
    _Float16* sH0   = sFeat + H0_OFF;               // [128][64]
    _Float16* sInp  = sFeat + INP_OFF;              // [128][288]
    _Float16* sH1   = sFeat + H1_OFF;               // [128][256]
    _Float16* sH2   = sFeat + H2_OFF;               // [128][256]
    float*    sDen  = (float*)(sFeat + LDS_HALVES); // [128]

    const int t    = threadIdx.x;
    const int lane = t & 31;
    const int wave = t >> 5;
    const int p    = blockIdx.x * BM + t;

    // --- Phase A: TDM load of the 128x64 f16 feature tile into LDS --------
    if (wave == 0) {
        unsigned long long ga = (unsigned long long)(uintptr_t)feats +
                                (unsigned long long)blockIdx.x * (BM * 64 * 2);
        // D# group 0: count=1, lds_addr=0, 57-bit global_addr, type=2
        u32x4 g0;
        g0.x = 1u;
        g0.y = 0u;
        g0.z = (unsigned)(ga & 0xFFFFFFFFull);
        g0.w = (unsigned)((ga >> 32) & 0x01FFFFFFull) | (2u << 30);
        // D# group 1: data_size=2B, tensor 64x128, tile 64x128, stride 64
        i32x8 g1;
        g1[0] = 0x00010000;            // data_size=1 (2 bytes), mask=0
        g1[1] = (int)(64u << 16);      // tensor_dim0 = 64 (bits 79:48)
        g1[2] = (int)(128u << 16);     // tensor_dim1 = 128 (bits 111:80)
        g1[3] = (int)(64u << 16);      // tile_dim0 = 64 (bits 127:112)
        g1[4] = 128;                   // tile_dim1 = 128 (bits 143:128)
        g1[5] = 64;                    // tensor_dim0_stride = 64 (bits 207:160)
        g1[6] = 0;
        g1[7] = 0;
        i32x4 z4 = {0, 0, 0, 0};
#if defined(__clang_major__) && (__clang_major__ >= 23)
        i32x8 z8 = {0, 0, 0, 0, 0, 0, 0, 0};
        __builtin_amdgcn_tensor_load_to_lds(g0, g1, z4, z4, z8, 0);
#else
        __builtin_amdgcn_tensor_load_to_lds(g0, g1, z4, z4, 0);
#endif
        __builtin_amdgcn_s_wait_tensorcnt(0);
    }
    // --- copy this point's direction encoding into the inp buffer ---------
    {
        const f32x4* s = (const f32x4*)(deG + (size_t)(p >> 5) * 32);
        f32x4* d = (f32x4*)(sInp + (size_t)t * 288 + 256);
#pragma unroll
        for (int i = 0; i < 4; ++i) d[i] = s[i];
    }
    __syncthreads();

    const int mlo = lane & 15;     // A row / C column selector
    const int hi  = lane >> 4;

    // ---------------- Layer 1: feats(64p) @ Wd0 -> relu -> h0(64) ---------
    {
        const v16h* B = (const v16h*)(wpack + OFF_WD0);
        for (int tile = wave; tile < 8 * 4; tile += NWAVES) {
            int mt = tile & 7, nt = tile >> 3;
            int m = mt * 16 + mlo, n = nt * 16 + mlo;
            v8f c = {};
            c = mm_accum(c, sFeat, 64, m, hi, B, nt * 2, 2, lane);
            float bias = bd0[n];
#pragma unroll
            for (int r = 0; r < 8; ++r) {
                float v = fmaxf(c[r] + bias, 0.0f);
                sH0[(size_t)(mt * 16 + hi * 8 + r) * 64 + n] = (_Float16)v;
            }
        }
    }
    __syncthreads();

    // ---------------- Layer 2: h0 @ Wd1 -> x(256) (+ density col) ---------
    {
        const v16h* B = (const v16h*)(wpack + OFF_WD1);
        for (int tile = wave; tile < 8 * 16; tile += NWAVES) {
            int mt = tile & 7, nt = tile >> 3;
            int m = mt * 16 + mlo, n = nt * 16 + mlo;
            v8f c = {};
            c = mm_accum(c, sH0, 64, m, hi, B, nt * 2, 2, lane);
            float bias = bd1[n];
#pragma unroll
            for (int r = 0; r < 8; ++r) {
                float v = c[r] + bias;
                sInp[(size_t)(mt * 16 + hi * 8 + r) * 288 + n] = (_Float16)v;
                if (nt == 0 && mlo == 0) {  // column 0 -> density
                    float tt = v - 1.0f;
                    sDen[mt * 16 + hi * 8 + r] =
                        (tt > 15.0f) ? tt : log1pf(expf(tt));
                }
            }
        }
    }
    __syncthreads();
    out[(size_t)p * 4 + 0] = sDen[t];

    // ---------------- Layer 3: inp(288p) @ W0 -> relu -> h1(256) ----------
    {
        const v16h* B = (const v16h*)(wpack + OFF_W0);
        for (int tile = wave; tile < 8 * 16; tile += NWAVES) {
            int mt = tile & 7, nt = tile >> 3;
            int m = mt * 16 + mlo, n = nt * 16 + mlo;
            v8f c = {};
            c = mm_accum(c, sInp, 288, m, hi, B, nt * 9, 9, lane);
            float bias = b0[n];
#pragma unroll
            for (int r = 0; r < 8; ++r) {
                float v = fmaxf(c[r] + bias, 0.0f);
                sH1[(size_t)(mt * 16 + hi * 8 + r) * 256 + n] = (_Float16)v;
            }
        }
    }
    __syncthreads();

    // ---------------- Layer 4: [h1,inp](544p) @ W1 -> relu -> h2(256) -----
    {
        const v16h* B = (const v16h*)(wpack + OFF_W1);
        for (int tile = wave; tile < 8 * 16; tile += NWAVES) {
            int mt = tile & 7, nt = tile >> 3;
            int m = mt * 16 + mlo, n = nt * 16 + mlo;
            v8f c = {};
            c = mm_accum(c, sH1, 256, m, hi, B, nt * 17, 8, lane);      // K 0..255
            c = mm_accum(c, sInp, 288, m, hi, B, nt * 17 + 8, 9, lane); // K 256..543
            float bias = b1[n];
#pragma unroll
            for (int r = 0; r < 8; ++r) {
                float v = fmaxf(c[r] + bias, 0.0f);
                sH2[(size_t)(mt * 16 + hi * 8 + r) * 256 + n] = (_Float16)v;
            }
        }
    }
    __syncthreads();

    // ---------------- Layer 5: h2 @ Wr -> sigmoid -> rgb ------------------
    {
        const v16h* B = (const v16h*)(wpack + OFF_WR);
        for (int mt = wave; mt < 8; mt += NWAVES) {
            int m = mt * 16 + mlo;
            v8f c = {};
            c = mm_accum(c, sH2, 256, m, hi, B, 0, 8, lane);
            if (mlo < 3) {
                float bias = br[mlo];
#pragma unroll
                for (int r = 0; r < 8; ++r) {
                    float v = c[r] + bias;
                    v = 1.0f / (1.0f + expf(-v));
                    v = v * 1.002f - 0.001f;
                    int row = mt * 16 + hi * 8 + r;
                    size_t pp = (size_t)blockIdx.x * BM + row;
                    out[pp * 4 + 1 + mlo] = v;
                }
            }
        }
    }
}

// ---------------------------------------------------------------------------
// Launch
// ---------------------------------------------------------------------------
extern "C" void kernel_launch(void* const* d_in, const int* in_sizes, int n_in,
                              void* d_out, int out_size, void* d_ws, size_t ws_size,
                              hipStream_t stream) {
    const float* coords   = (const float*)d_in[0];
    const float* viewdirs = (const float*)d_in[1];
    const float* table    = (const float*)d_in[2];
    const float* Wd0 = (const float*)d_in[3];
    const float* bd0 = (const float*)d_in[4];
    const float* Wd1 = (const float*)d_in[5];
    const float* bd1 = (const float*)d_in[6];
    const float* W0  = (const float*)d_in[7];
    const float* b0  = (const float*)d_in[8];
    const float* W1  = (const float*)d_in[9];
    const float* b1  = (const float*)d_in[10];
    const float* Wr  = (const float*)d_in[11];
    const float* br  = (const float*)d_in[12];
    _Float16* wp = (_Float16*)d_ws;

    auto pack = [&](const float* W, int Kr, int Nr, int Kp, int Np, int off) {
        int total = Kp * Np;
        pack_weights<<<(total + 255) / 256, 256, 0, stream>>>(W, Kr, Nr, Kp, Np,
                                                              wp + off);
    };
    pack(Wd0,  40,  64,  64,  64, OFF_WD0);
    pack(Wd1,  64, 256,  64, 256, OFF_WD1);
    pack(W0,  283, 256, 288, 256, OFF_W0);
    pack(W1,  539, 256, 544, 256, OFF_W1);
    pack(Wr,  256,   3, 256,  16, OFF_WR);

    grid_encode_kernel<<<NPTS / 256, 256, 0, stream>>>(coords, table,
                                                       wp + OFF_FEATS);
    direnc_kernel<<<NRAYS / 256, 256, 0, stream>>>(viewdirs, wp + OFF_DE);

    nerf_mlp<<<NPTS / BM, NTHREADS, SMEM_BYTES, stream>>>(
        wp + OFF_FEATS, wp + OFF_DE, wp, bd0, bd1, b0, b1, br, (float*)d_out);
}
